// LocalAttention_73675868995958
// MI455X (gfx1250) — matrix operand order repaired and checked by reading, not compiled, not run on hardware
//
#include <hip/hip_runtime.h>

// ---------------------------------------------------------------------------
// LocalAttention forward for MI455X (gfx1250, wave32, WMMA).
// Pipeline: f32->bf16 convert (+weight transpose), QKV GEMM(+bias),
// per-chunk Q*K^T GEMM, masked softmax, V transpose, P*V GEMM,
// output-projection GEMM(+bias).
// Matrix math: v_wmma_f32_16x16x32_bf16 (fp32 accumulate).
// All GEMM operands staged via async global->LDS (ASYNCcnt), double-buffered;
// every B operand is pre-transposed to [N][K] so staging is pure b128 copies.
// ---------------------------------------------------------------------------

typedef __attribute__((ext_vector_type(16))) __bf16 v16bf;
typedef __attribute__((ext_vector_type(8)))  float  v8f;
typedef int v4i __attribute__((vector_size(16)));   // matches builtin param type

#if defined(__has_builtin)
#if __has_builtin(__builtin_amdgcn_global_load_async_to_lds_b128) && \
    __has_builtin(__builtin_amdgcn_s_wait_asynccnt)
#define HAVE_ASYNC_LDS 1
#endif
#endif
#ifndef HAVE_ASYNC_LDS
#define HAVE_ASYNC_LDS 0
#endif

#if HAVE_ASYNC_LDS
__device__ __forceinline__ void async_cp16(const unsigned short* g, unsigned short* l) {
    // 16-byte async copy, global -> LDS, bypassing VGPRs (tracked by ASYNCcnt)
    __builtin_amdgcn_global_load_async_to_lds_b128(
        (__attribute__((address_space(1))) v4i*)g,
        (__attribute__((address_space(3))) v4i*)l,
        0, 0);
}
#endif

__device__ __forceinline__ unsigned short f2bf(float f) {
    unsigned int u = __float_as_uint(f);
    u += 0x7FFFu + ((u >> 16) & 1u);   // round-to-nearest-even
    return (unsigned short)(u >> 16);
}

// ---------------------------------------------------------------------------
// Elementwise fp32 -> bf16 (vectorized x4)
// ---------------------------------------------------------------------------
__global__ __launch_bounds__(256) void convert_f32_bf16(
    const float* __restrict__ src, unsigned short* __restrict__ dst, int n4)
{
    int i = blockIdx.x * 256 + threadIdx.x;
    if (i >= n4) return;
    float4 f = ((const float4*)src)[i];
    union { unsigned short s[4]; uint2 u; } o;
    o.s[0] = f2bf(f.x); o.s[1] = f2bf(f.y);
    o.s[2] = f2bf(f.z); o.s[3] = f2bf(f.w);
    ((uint2*)dst)[i] = o.u;
}

// ---------------------------------------------------------------------------
// Transposing fp32 -> bf16 convert: src [rows][cols] f32 -> dst [cols][rows]
// bf16. 32x32 LDS tile, both global sides coalesced. rows, cols % 32 == 0.
// ---------------------------------------------------------------------------
__global__ __launch_bounds__(256) void transpose_conv_f32_bf16(
    const float* __restrict__ src, unsigned short* __restrict__ dst,
    int rows, int cols)
{
    __shared__ unsigned short t[32][33];
    const int r0 = blockIdx.y * 32;
    const int c0 = blockIdx.x * 32;
    const int tx = threadIdx.x & 31;
    const int ty = threadIdx.x >> 5;    // 0..7
#pragma unroll
    for (int i = 0; i < 4; ++i) {
        const int r = r0 + ty + i * 8;
        t[ty + i * 8][tx] = f2bf(src[(long long)r * cols + c0 + tx]);
    }
    __syncthreads();
#pragma unroll
    for (int i = 0; i < 4; ++i) {
        const int c = c0 + ty + i * 8;  // row of dst
        dst[(long long)c * rows + r0 + tx] = t[tx][ty + i * 8];
    }
}

// ---------------------------------------------------------------------------
// Per-chunk V transpose (bf16): vt[z][d][c] = qkv[(z*512+c)*3072 + 2048 + d]
// 32x32 LDS tile. grid = (1024/32, 512/32, 64).
// ---------------------------------------------------------------------------
__global__ __launch_bounds__(256) void transpose_v_bf16(
    const unsigned short* __restrict__ qkv, unsigned short* __restrict__ vt)
{
    __shared__ unsigned short t[32][33];
    const int d0 = blockIdx.x * 32;
    const int c0 = blockIdx.y * 32;
    const long long z = blockIdx.z;
    const int tx = threadIdx.x & 31;
    const int ty = threadIdx.x >> 5;
#pragma unroll
    for (int i = 0; i < 4; ++i) {
        const int c = c0 + ty + i * 8;
        t[ty + i * 8][tx] = qkv[(z * 512 + c) * 3072 + 2048 + d0 + tx];
    }
    __syncthreads();
#pragma unroll
    for (int i = 0; i < 4; ++i) {
        const int d = d0 + ty + i * 8;
        vt[z * 524288 + (long long)d * 512 + c0 + tx] = t[tx][ty + i * 8];
    }
}

// ---------------------------------------------------------------------------
// WMMA bf16 GEMM:  C[z] = A[z] (M x K, [m][k]) * Bt[z]^T  (+bias)
//   Bt stored [N][K] row-major (pre-transposed B) -> staging is contiguous.
//   OutT = unsigned short (bf16 raw) or float.
// Macro-tile 128x128, K-step 32, double-buffered LDS, fully async staging.
// M,N multiples of 128; K multiple of 32 (true for all uses here).
// ---------------------------------------------------------------------------
template <typename OutT>
__global__ __launch_bounds__(256) void wmma_gemm_bf16(
    const unsigned short* __restrict__ A,  int lda, long long strideAz,
    const unsigned short* __restrict__ Bt, int ldb, long long strideBz,
    OutT* __restrict__ C, int ldc, long long strideCz,
    const float* __restrict__ bias, int K)
{
    __shared__ unsigned short As[2][128][32];   // [buf][m][k]
    __shared__ unsigned short Bs[2][128][32];   // [buf][n][k]

    const int tid  = threadIdx.x;
    const int lane = tid & 31;
    const int wave = tid >> 5;
    const int wm   = wave & 3;   // wave row -> 32 rows
    const int wn   = wave >> 2;  // wave col -> 64 cols

    const int bm = blockIdx.y * 128;
    const int bn = blockIdx.x * 128;
    const long long z = blockIdx.z;

    A  += z * strideAz;
    Bt += z * strideBz;
    C  += z * strideCz;

    v8f acc[2][4];
#pragma unroll
    for (int i = 0; i < 2; ++i)
#pragma unroll
        for (int j = 0; j < 4; ++j) acc[i][j] = (v8f)0.0f;

    // staging coordinates (all dims exact multiples -> no guards)
    const int srow = tid & 127;                // 0..127
    const int scol = (tid >> 7) << 4;          // 0 or 16

    auto stage = [&](int buf, int k0) {
        const unsigned short* sa = A  + (long long)(bm + srow) * lda + k0 + scol;
        const unsigned short* sb = Bt + (long long)(bn + srow) * ldb + k0 + scol;
#if HAVE_ASYNC_LDS
        async_cp16(sa,     &As[buf][srow][scol]);
        async_cp16(sa + 8, &As[buf][srow][scol + 8]);
        async_cp16(sb,     &Bs[buf][srow][scol]);
        async_cp16(sb + 8, &Bs[buf][srow][scol + 8]);
#else
        *(uint4*)&As[buf][srow][scol]     = *(const uint4*)(sa);
        *(uint4*)&As[buf][srow][scol + 8] = *(const uint4*)(sa + 8);
        *(uint4*)&Bs[buf][srow][scol]     = *(const uint4*)(sb);
        *(uint4*)&Bs[buf][srow][scol + 8] = *(const uint4*)(sb + 8);
#endif
    };

    const int nk = K >> 5;          // K / 32 tiles
    stage(0, 0);

    for (int t = 0; t < nk; ++t) {
        const int cur = t & 1;

        if (t + 1 < nk) {
            stage(cur ^ 1, (t + 1) << 5);
            if (t + 2 < nk) {   // hint tile t+2 toward the caches
                const int kp = (t + 2) << 5;
                __builtin_prefetch(A  + (long long)(bm + srow) * lda + kp + scol, 0, 3);
                __builtin_prefetch(Bt + (long long)(bn + srow) * ldb + kp + scol, 0, 3);
            }
#if HAVE_ASYNC_LDS
            __builtin_amdgcn_s_wait_asynccnt(4);    // current tile resident
#endif
        } else {
#if HAVE_ASYNC_LDS
            __builtin_amdgcn_s_wait_asynccnt(0);
#endif
        }
        __syncthreads();

        // ---- fragments (CDNA5 wave32 layouts) and 8 WMMAs
        const int l16   = lane & 15;
        const int hi    = lane >> 4;          // 0 = lanes 0-15, 1 = lanes 16-31
        const int koffA = hi << 3;            // A: K 0-7/16-23 vs 8-15/24-31
        const int koffB = hi << 3;            // B: K 0-7/16-23 vs 8-15/24-31

        union { v16bf v; uint4 u[2]; } fa[2], fb[4];
#pragma unroll
        for (int mi = 0; mi < 2; ++mi) {
            const int r = wm * 32 + mi * 16 + l16;
            fa[mi].u[0] = *(const uint4*)&As[cur][r][koffA];
            fa[mi].u[1] = *(const uint4*)&As[cur][r][16 + koffA];
        }
#pragma unroll
        for (int ni = 0; ni < 4; ++ni) {
            const int c = wn * 64 + ni * 16 + l16;
            fb[ni].u[0] = *(const uint4*)&Bs[cur][c][koffB];
            fb[ni].u[1] = *(const uint4*)&Bs[cur][c][16 + koffB];
        }
#pragma unroll
        for (int mi = 0; mi < 2; ++mi)
#pragma unroll
            for (int ni = 0; ni < 4; ++ni)
                acc[mi][ni] = __builtin_amdgcn_wmma_f32_16x16x32_bf16(
                    false, fa[mi].v, false, fb[ni].v,
                    (short)0, acc[mi][ni], false, false);
        __syncthreads();
    }

    // ---- store C (+bias). VGPR r <-> M = base + r + 8*(lane>=16); N = lane%16
    const int l16 = lane & 15;
    const int hi8 = (lane >> 4) << 3;
#pragma unroll
    for (int mi = 0; mi < 2; ++mi) {
#pragma unroll
        for (int ni = 0; ni < 4; ++ni) {
            const int n  = bn + wn * 64 + ni * 16 + l16;
            const int m0 = bm + wm * 32 + mi * 16 + hi8;
            const float bv = bias ? bias[n] : 0.0f;
#pragma unroll
            for (int r = 0; r < 8; ++r) {
                const float v = acc[mi][ni][r] + bv;
                const long long idx = (long long)(m0 + r) * ldc + n;
                if constexpr (sizeof(OutT) == 2)
                    C[idx] = (OutT)f2bf(v);
                else
                    C[idx] = (OutT)v;
            }
        }
    }
}

// ---------------------------------------------------------------------------
// Row softmax with causal mask, scale 1/sqrt(1024) = 1/32. One block per row
// of 512 scores; writes bf16 probabilities (zeros beyond causal length).
// ---------------------------------------------------------------------------
__global__ __launch_bounds__(256) void softmax_causal(
    const float* __restrict__ scores, unsigned short* __restrict__ attn,
    const int* __restrict__ maskp)
{
    __shared__ float red[256];
    const int row = blockIdx.x;            // z*512 + r
    const int r   = row & 511;
    const float* s = scores + (long long)row * 512;
    unsigned short* p = attn + (long long)row * 512;
    const int len = (maskp[0] != 0) ? (r + 1) : 512;
    const float scale = 0.03125f;
    const int tid = threadIdx.x;

    float m = -3.4e38f;
    for (int j = tid; j < len; j += 256) m = fmaxf(m, s[j] * scale);
    red[tid] = m; __syncthreads();
    for (int off = 128; off; off >>= 1) {
        if (tid < off) red[tid] = fmaxf(red[tid], red[tid + off]);
        __syncthreads();
    }
    m = red[0]; __syncthreads();

    float sum = 0.0f;
    for (int j = tid; j < len; j += 256) sum += __expf(s[j] * scale - m);
    red[tid] = sum; __syncthreads();
    for (int off = 128; off; off >>= 1) {
        if (tid < off) red[tid] += red[tid + off];
        __syncthreads();
    }
    const float inv = 1.0f / red[0];

    for (int j = tid; j < 512; j += 256) {
        const float v = (j < len) ? __expf(s[j] * scale - m) * inv : 0.0f;
        p[j] = f2bf(v);
    }
}

// ---------------------------------------------------------------------------
// Launcher. Shapes: B=8, S=4096, Din=Dout=1024, H=1, n_chunks=8, C=512.
// ---------------------------------------------------------------------------
extern "C" void kernel_launch(void* const* d_in, const int* in_sizes, int n_in,
                              void* d_out, int out_size, void* d_ws, size_t ws_size,
                              hipStream_t stream)
{
    (void)in_sizes; (void)n_in; (void)out_size; (void)ws_size;

    const float* x     = (const float*)d_in[0];   // [32768, 1024]
    const float* W_qkv = (const float*)d_in[1];   // [1024, 3072]
    const float* b_qkv = (const float*)d_in[2];   // [3072]
    const float* W_out = (const float*)d_in[3];   // [1024, 1024]
    const float* b_out = (const float*)d_in[4];   // [1024]
    const int*   mask  = (const int*)d_in[5];     // scalar
    float*       out   = (float*)d_out;           // [32768, 1024]

    const int M   = 32768;   // B*S
    const int Din = 1024, N3 = 3072, D = 1024;
    const int NC  = 64;      // B * n_chunks
    const int Cck = 512;     // chunk length

    // workspace carve-up (bf16 stored as raw u16)
    char* ws = (char*)d_ws;
    unsigned short* x_bf    = (unsigned short*)(ws);                   // 64 MB
    unsigned short* wqkv_t  = (unsigned short*)(ws + 67108864LL);      //  6 MB [3072][1024]
    unsigned short* wout_t  = (unsigned short*)(ws + 73400320LL);      //  2 MB [1024][1024]
    unsigned short* qkv_b   = (unsigned short*)(ws + 75497472LL);      // 192 MB
    float*          scores  = (float*)(ws + 276824064LL);              // 64 MB
    unsigned short* v_t     = (unsigned short*)(ws + 276824064LL);     // reuse after softmax
    unsigned short* attn_b  = (unsigned short*)(ws + 343932928LL);     // 32 MB
    unsigned short* o_bf    = (unsigned short*)(ws + 377487360LL);     // 64 MB

    // 1) fp32 -> bf16 conversions; weights transposed to [N][K]
    convert_f32_bf16<<<(M * Din / 4 + 255) / 256, 256, 0, stream>>>(x, x_bf, M * Din / 4);
    transpose_conv_f32_bf16<<<dim3(N3 / 32, Din / 32), 256, 0, stream>>>(W_qkv, wqkv_t, Din, N3);
    transpose_conv_f32_bf16<<<dim3(D / 32, D / 32), 256, 0, stream>>>(W_out, wout_t, D, D);

    // 2) QKV projection: [32768,1024] x [1024,3072] + bias -> bf16
    wmma_gemm_bf16<unsigned short>
        <<<dim3(N3 / 128, M / 128, 1), 256, 0, stream>>>(
            x_bf, Din, 0LL, wqkv_t, Din, 0LL, qkv_b, N3, 0LL, b_qkv, Din);

    // 3) scores[z] = Q[z] * K[z]^T  (K rows already give [N][K] layout)
    wmma_gemm_bf16<float>
        <<<dim3(Cck / 128, Cck / 128, NC), 256, 0, stream>>>(
            qkv_b + 0,    N3, (long long)Cck * N3,
            qkv_b + 1024, N3, (long long)Cck * N3,
            scores, Cck, (long long)Cck * Cck,
            nullptr, D);

    // 4) causal softmax -> bf16 probabilities
    softmax_causal<<<NC * Cck, 256, 0, stream>>>(scores, attn_b, mask);

    // 5) transpose V per chunk into the (now dead) scores region: [z][d][c]
    transpose_v_bf16<<<dim3(D / 32, Cck / 32, NC), 256, 0, stream>>>(qkv_b, v_t);

    //    O[z] = P[z] * V[z]  via V^T tiles -> bf16
    wmma_gemm_bf16<unsigned short>
        <<<dim3(D / 128, Cck / 128, NC), 256, 0, stream>>>(
            attn_b, Cck, (long long)Cck * Cck,
            v_t, Cck, (long long)Cck * D,
            o_bf, D, (long long)Cck * D,
            nullptr, Cck);

    // 6) output projection: [32768,1024] x [1024,1024] + bias -> fp32 d_out
    wmma_gemm_bf16<float>
        <<<dim3(D / 128, M / 128, 1), 256, 0, stream>>>(
            o_bf, D, 0LL, wout_t, D, 0LL, out, D, 0LL, b_out, D);
}
